// ie_HGCN_28011776704703
// MI455X (gfx1250) — compile-verified, matrix operand order checked by reading, not run on hardware
//
#include <hip/hip_runtime.h>
#include <hip/hip_bf16.h>

typedef __attribute__((ext_vector_type(16))) __bf16 bf16x16;
typedef __attribute__((ext_vector_type(8)))  float  f32x8;

union FragU { bf16x16 v; uint4 q[2]; };

__device__ __forceinline__ unsigned int f2bf(float f) {
  unsigned int u = __float_as_uint(f);
  u += 0x7fffu + ((u >> 16) & 1u);
  return u >> 16;
}
__device__ __forceinline__ float eluf(float x) { return x > 0.f ? x : expm1f(x); }

__device__ __forceinline__ void wait_async_4() {
#if __has_builtin(__builtin_amdgcn_s_wait_asynccnt)
  __builtin_amdgcn_s_wait_asynccnt(4);
#else
  asm volatile("s_wait_asynccnt 0x4" ::: "memory");
#endif
}
__device__ __forceinline__ void wait_async_0() {
#if __has_builtin(__builtin_amdgcn_s_wait_asynccnt)
  __builtin_amdgcn_s_wait_asynccnt(0);
#else
  asm volatile("s_wait_asynccnt 0x0" ::: "memory");
#endif
}
// Async copy 16B from global to LDS (per lane). ASYNCcnt tracked.
__device__ __forceinline__ void async_copy_b128(void* lds_dst, const void* gsrc) {
  unsigned lds = (unsigned)(unsigned long long)lds_dst;          // LDS offset (addr[31:0])
  unsigned long long ga = (unsigned long long)gsrc;
  asm volatile("global_load_async_to_lds_b128 %0, %1, off" :: "v"(lds), "v"(ga) : "memory");
}

// ---------------------------------------------------------------------------
// f32 -> bf16 conversion (4 elems per thread)
// ---------------------------------------------------------------------------
__global__ void cvt_bf16(const float4* __restrict__ src, uint2* __restrict__ dst, long n4) {
  long i = (long)blockIdx.x * blockDim.x + threadIdx.x;
  if (i >= n4) return;
  float4 f = src[i];
  uint2 o;
  o.x = f2bf(f.x) | (f2bf(f.y) << 16);
  o.y = f2bf(f.z) | (f2bf(f.w) << 16);
  dst[i] = o;
}

__global__ void zero_f32(float4* __restrict__ p, long n4) {
  long i = (long)blockIdx.x * blockDim.x + threadIdx.x;
  if (i < n4) p[i] = make_float4(0.f, 0.f, 0.f, 0.f);
}

// ---------------------------------------------------------------------------
// vk[lt,d] = sum_a Wk[lt,a,d]*wv_k[a];  vq[lt,d] = sum_a Wq[lt,a,d]*wv_q[a]
// ---------------------------------------------------------------------------
__global__ void make_vkvq(const float* __restrict__ Wq, const float* __restrict__ Wk,
                          const float* __restrict__ Wv,
                          float* __restrict__ vk, float* __restrict__ vq) {
  int idx = blockIdx.x * 256 + threadIdx.x;
  if (idx >= 2 * 3 * 256) return;
  int d = idx & 255, lt = idx >> 8;
  const float* wkp = Wk + (long)lt * 128 * 256;
  const float* wqp = Wq + (long)lt * 128 * 256;
  const float* wvp = Wv + (long)lt * 256;
  float sk = 0.f, sq = 0.f;
  for (int a = 0; a < 128; ++a) {
    sk += wkp[(long)a * 256 + d] * wvp[a];
    sq += wqp[(long)a * 256 + d] * wvp[128 + a];
  }
  vk[idx] = sk; vq[idx] = sq;
}

// ---------------------------------------------------------------------------
// bf16 WMMA GEMM:  C[M,N] (f32) = A[M,K] bf16 row-major * B[N,K]^T bf16 row-major
// Block: 256 threads = 8 waves. Tile 128(M) x 128(N). K stepped by 32.
// Double-buffered LDS staged with GLOBAL_LOAD_ASYNC_TO_LDS_B128 (ASYNCcnt).
// Per wave: 32x64 output = 2x4 WMMA 16x16 tiles.
// Out-of-range M/N rows are CLAMPED (not zeroed): a clamped A row only feeds
// C rows that fail the store mask, ditto clamped B rows for C columns.
// ---------------------------------------------------------------------------
__global__ __launch_bounds__(256) void gemm_bf16_wmma(
    const uint4* __restrict__ A, const uint4* __restrict__ B,
    float* __restrict__ C, int M, int N, int K) {
  __shared__ uint4 sA[2][512];  // 128 rows x 32 bf16 (4 uint4/row) per stage
  __shared__ uint4 sB[2][512];  // 128 rows x 32 bf16 per stage
  const int tid = threadIdx.x;
  const int wave = tid >> 5, lane = tid & 31;
  const int wm = wave & 3, wn = wave >> 2;
  const int m0 = blockIdx.y * 128;
  const int n0 = blockIdx.x * 128;
  const int K8 = K >> 3;
  const int hf = lane >> 4, r16 = lane & 15;
  const int nsteps = K >> 5;

  // per-thread staging coordinates: thread covers chunks tid and tid+256 of each tile
  const int r0 = tid >> 2, w0 = tid & 3;          // chunk tid
  const int r1 = (tid + 256) >> 2, w1 = tid & 3;  // chunk tid+256
  const long gmA0 = (m0 + r0 < M) ? (m0 + r0) : (M - 1);
  const long gmA1 = (m0 + r1 < M) ? (m0 + r1) : (M - 1);
  const long gnB0 = (n0 + r0 < N) ? (n0 + r0) : (N - 1);
  const long gnB1 = (n0 + r1 < N) ? (n0 + r1) : (N - 1);

  f32x8 acc[2][4] = {};

  auto issue = [&](int kq, int st) {
    async_copy_b128(&sA[st][tid],       &A[gmA0 * K8 + kq + w0]);
    async_copy_b128(&sA[st][tid + 256], &A[gmA1 * K8 + kq + w1]);
    async_copy_b128(&sB[st][tid],       &B[gnB0 * K8 + kq + w0]);
    async_copy_b128(&sB[st][tid + 256], &B[gnB1 * K8 + kq + w1]);
  };

  issue(0, 0);
  for (int ks = 0; ks < nsteps; ++ks) {
    const int st = ks & 1;
    if (ks + 1 < nsteps) {
      issue((ks + 1) << 2, st ^ 1);
      wait_async_4();           // stage-ks loads (issued earlier) retired, in order
    } else {
      wait_async_0();
    }
    __syncthreads();            // all waves' stage data visible in LDS

    FragU af[2], bfr[4];
#pragma unroll
    for (int s = 0; s < 2; ++s) {
      int r = wm * 32 + s * 16 + r16;
      af[s].q[0] = sA[st][r * 4 + hf];
      af[s].q[1] = sA[st][r * 4 + 2 + hf];
    }
#pragma unroll
    for (int t = 0; t < 4; ++t) {
      int r = wn * 64 + t * 16 + r16;
      bfr[t].q[0] = sB[st][r * 4 + hf];
      bfr[t].q[1] = sB[st][r * 4 + 2 + hf];
    }
#pragma unroll
    for (int s = 0; s < 2; ++s)
#pragma unroll
      for (int t = 0; t < 4; ++t)
        acc[s][t] = __builtin_amdgcn_wmma_f32_16x16x32_bf16(
            false, af[s].v, false, bfr[t].v, (short)0, acc[s][t], false, false);
    __syncthreads();            // all waves done reading this stage buffer
  }

  // C layout: lane holds col n = lane&15; VGPR r -> row r + 8*(lane>>4)
#pragma unroll
  for (int s = 0; s < 2; ++s)
#pragma unroll
    for (int t = 0; t < 4; ++t) {
      int n = n0 + wn * 64 + t * 16 + r16;
      if (n < N) {
#pragma unroll
        for (int r = 0; r < 8; ++r) {
          int m = m0 + wm * 32 + s * 16 + r + 8 * hf;
          if (m < M) C[(long)m * N + n] = acc[s][t][r];
        }
      }
    }
}

// ---------------------------------------------------------------------------
// SpMM scatter: Z[row[e], :] += T[col[e], :] * val[e]   (D=256, one wave/edge)
// ---------------------------------------------------------------------------
__global__ void spmm_scatter(const int* __restrict__ rows, const int* __restrict__ cols,
                             const float* __restrict__ val, const float* __restrict__ T,
                             float* __restrict__ Z, int E) {
  int e = blockIdx.x * 8 + (threadIdx.x >> 5);
  if (e >= E) return;
  int lane = threadIdx.x & 31;
  int r = rows[e], c = cols[e];
  float v = val[e];
  const float4* tp = (const float4*)(T + (long)c * 256 + lane * 8);
  float4 t0 = tp[0], t1 = tp[1];
  float* zp = Z + (long)r * 256 + lane * 8;
  atomicAdd(zp + 0, t0.x * v); atomicAdd(zp + 1, t0.y * v);
  atomicAdd(zp + 2, t0.z * v); atomicAdd(zp + 3, t0.w * v);
  atomicAdd(zp + 4, t1.x * v); atomicAdd(zp + 5, t1.y * v);
  atomicAdd(zp + 6, t1.z * v); atomicAdd(zp + 7, t1.w * v);
}

// ---------------------------------------------------------------------------
// Per-node combine: e_r = elu(z_r·vk + z_self·vq); softmax over rels;
// h = elu(sum a_r z_r). Writes bf16 mirror (+ optional f32 emb). 1 wave/node.
// ---------------------------------------------------------------------------
__global__ void combine_kernel(const float* __restrict__ zs,
                               const float* __restrict__ zr0,
                               const float* __restrict__ zr1, int nrel,
                               const float* __restrict__ vk, const float* __restrict__ vq,
                               uint4* __restrict__ hbf, float* __restrict__ emb, int N) {
  int node = blockIdx.x * 8 + (threadIdx.x >> 5);
  if (node >= N) return;
  int lane = threadIdx.x & 31;
  long base = (long)node * 256 + lane * 8;
  float z0[8], z1[8], z2[8], kv[8], qv[8];
  {
    const float4* p = (const float4*)(zs + base);
    float4 a = p[0], b = p[1];
    z0[0]=a.x; z0[1]=a.y; z0[2]=a.z; z0[3]=a.w; z0[4]=b.x; z0[5]=b.y; z0[6]=b.z; z0[7]=b.w;
  }
  {
    const float4* p = (const float4*)(zr0 + base);
    float4 a = p[0], b = p[1];
    z1[0]=a.x; z1[1]=a.y; z1[2]=a.z; z1[3]=a.w; z1[4]=b.x; z1[5]=b.y; z1[6]=b.z; z1[7]=b.w;
  }
  if (nrel > 1) {
    const float4* p = (const float4*)(zr1 + base);
    float4 a = p[0], b = p[1];
    z2[0]=a.x; z2[1]=a.y; z2[2]=a.z; z2[3]=a.w; z2[4]=b.x; z2[5]=b.y; z2[6]=b.z; z2[7]=b.w;
  } else {
#pragma unroll
    for (int i = 0; i < 8; ++i) z2[i] = 0.f;
  }
#pragma unroll
  for (int i = 0; i < 8; ++i) { kv[i] = vk[lane * 8 + i]; qv[i] = vq[lane * 8 + i]; }
  float pk = 0.f, pq = 0.f, p0 = 0.f, p1 = 0.f;
#pragma unroll
  for (int i = 0; i < 8; ++i) {
    pk += z0[i] * kv[i]; pq += z0[i] * qv[i];
    p0 += z1[i] * kv[i]; p1 += z2[i] * kv[i];
  }
#pragma unroll
  for (int off = 16; off > 0; off >>= 1) {
    pk += __shfl_xor(pk, off, 32);
    pq += __shfl_xor(pq, off, 32);
    p0 += __shfl_xor(p0, off, 32);
    p1 += __shfl_xor(p1, off, 32);
  }
  float e0 = eluf(pk + pq), e1 = eluf(p0 + pq);
  float e2 = (nrel > 1) ? eluf(p1 + pq) : -3.0e38f;
  float mx = fmaxf(fmaxf(e0, e1), e2);
  float a0 = expf(e0 - mx), a1 = expf(e1 - mx);
  float a2 = (nrel > 1) ? expf(e2 - mx) : 0.f;
  float inv = 1.f / (a0 + a1 + a2);
  a0 *= inv; a1 *= inv; a2 *= inv;
  float o[8];
#pragma unroll
  for (int i = 0; i < 8; ++i) o[i] = eluf(a0 * z0[i] + a1 * z1[i] + a2 * z2[i]);
  uint4 pk4;
  pk4.x = f2bf(o[0]) | (f2bf(o[1]) << 16);
  pk4.y = f2bf(o[2]) | (f2bf(o[3]) << 16);
  pk4.z = f2bf(o[4]) | (f2bf(o[5]) << 16);
  pk4.w = f2bf(o[6]) | (f2bf(o[7]) << 16);
  hbf[(long)node * 32 + lane] = pk4;
  if (emb) {
    float4* ep = (float4*)(emb + base);
    ep[0] = make_float4(o[0], o[1], o[2], o[3]);
    ep[1] = make_float4(o[4], o[5], o[6], o[7]);
  }
}

// ---------------------------------------------------------------------------
// In-place row softmax (C cols per row, one 256-thread block per row)
// ---------------------------------------------------------------------------
__global__ void row_softmax(float* __restrict__ P, int C) {
  __shared__ float red[256];
  float* p = P + (long)blockIdx.x * C;
  int t = threadIdx.x;
  float m = -3.0e38f;
  for (int c = t; c < C; c += 256) m = fmaxf(m, p[c]);
  red[t] = m; __syncthreads();
  for (int s = 128; s > 0; s >>= 1) { if (t < s) red[t] = fmaxf(red[t], red[t + s]); __syncthreads(); }
  m = red[0]; __syncthreads();
  float sum = 0.f;
  for (int c = t; c < C; c += 256) sum += expf(p[c] - m);
  red[t] = sum; __syncthreads();
  for (int s = 128; s > 0; s >>= 1) { if (t < s) red[t] += red[t + s]; __syncthreads(); }
  float inv = 1.f / red[0]; __syncthreads();
  for (int c = t; c < C; c += 256) p[c] = expf(p[c] - m) * inv;
}

// ---------------------------------------------------------------------------
extern "C" void kernel_launch(void* const* d_in, const int* in_sizes, int n_in,
                              void* d_out, int out_size, void* d_ws, size_t ws_size,
                              hipStream_t stream) {
  (void)in_sizes; (void)n_in; (void)out_size; (void)ws_size;
  const float* x_A   = (const float*)d_in[0];
  const float* x_P   = (const float*)d_in[1];
  const float* x_C   = (const float*)d_in[2];
  const float* val_ap = (const float*)d_in[3];
  const float* val_pa = (const float*)d_in[4];
  const float* val_pc = (const float*)d_in[5];
  const float* val_cp = (const float*)d_in[6];
  const float* W_self = (const float*)d_in[7];
  const float* W_sem  = (const float*)d_in[8];
  const float* Wq     = (const float*)d_in[9];
  const float* Wk     = (const float*)d_in[10];
  const float* Wv     = (const float*)d_in[11];
  const float* W_cls  = (const float*)d_in[12];
  const int* ap_row = (const int*)d_in[13];
  const int* ap_col = (const int*)d_in[14];
  const int* pc_row = (const int*)d_in[15];
  const int* pc_col = (const int*)d_in[16];

  constexpr int NA = 50000, NP = 100000, NC = 2000, D = 256, LNUM = 2;
  constexpr int EAP = 600000, EPC = 100000, NCLS = 2000;

  char* w = (char*)d_ws;
  size_t off = 0;
  auto alloc = [&](size_t bytes) -> char* {
    char* p = w + off;
    off += (bytes + 255) & ~(size_t)255;
    return p;
  };
  unsigned short* hbfA   = (unsigned short*)alloc((size_t)NA * D * 2);
  unsigned short* hbfP   = (unsigned short*)alloc((size_t)NP * D * 2);
  unsigned short* hbfC   = (unsigned short*)alloc((size_t)NC * D * 2);
  unsigned short* WselfB = (unsigned short*)alloc((size_t)LNUM * 3 * D * D * 2);
  unsigned short* WsemB  = (unsigned short*)alloc((size_t)LNUM * 4 * D * D * 2);
  unsigned short* WclsB  = (unsigned short*)alloc((size_t)NCLS * D * 2);
  float* vkb    = (float*)alloc((size_t)LNUM * 3 * D * 4);
  float* vqb    = (float*)alloc((size_t)LNUM * 3 * D * 4);
  float* zselfA = (float*)alloc((size_t)NA * D * 4);
  float* zselfP = (float*)alloc((size_t)NP * D * 4);
  float* zselfC = (float*)alloc((size_t)NC * D * 4);
  float* zrelA  = (float*)alloc((size_t)NA * D * 4);
  float* zrelP0 = (float*)alloc((size_t)NP * D * 4);
  float* zrelP1 = (float*)alloc((size_t)NP * D * 4);
  float* zrelC  = (float*)alloc((size_t)NC * D * 4);
  float* tbuf   = (float*)alloc((size_t)NP * D * 4);

  auto cvt = [&](const float* src, unsigned short* dst, long n) {
    long n4 = n / 4;
    cvt_bf16<<<(unsigned)((n4 + 255) / 256), 256, 0, stream>>>((const float4*)src, (uint2*)dst, n4);
  };
  auto gemm = [&](const unsigned short* A, const unsigned short* B, float* C, int M, int N) {
    dim3 g((unsigned)((N + 127) / 128), (unsigned)((M + 127) / 128));
    gemm_bf16_wmma<<<g, 256, 0, stream>>>((const uint4*)A, (const uint4*)B, C, M, N, 256);
  };
  auto zero = [&](float* p, long n) {
    long n4 = n / 4;
    zero_f32<<<(unsigned)((n4 + 255) / 256), 256, 0, stream>>>((float4*)p, n4);
  };
  auto scat = [&](const int* r, const int* c, const float* v, const float* T, float* Z, int E) {
    spmm_scatter<<<(unsigned)((E + 7) / 8), 256, 0, stream>>>(r, c, v, T, Z, E);
  };
  auto comb = [&](const float* zs, const float* z0, const float* z1, int nrel, int lt,
                  unsigned short* hb, float* emb, int N) {
    combine_kernel<<<(unsigned)((N + 7) / 8), 256, 0, stream>>>(
        zs, z0, z1, nrel, vkb + (size_t)lt * D, vqb + (size_t)lt * D, (uint4*)hb, emb, N);
  };

  // bf16 mirrors of activations + weights; attention vectors
  cvt(x_A, hbfA, (long)NA * D);
  cvt(x_P, hbfP, (long)NP * D);
  cvt(x_C, hbfC, (long)NC * D);
  cvt(W_self, WselfB, (long)LNUM * 3 * D * D);
  cvt(W_sem, WsemB, (long)LNUM * 4 * D * D);
  cvt(W_cls, WclsB, (long)NCLS * D);
  make_vkvq<<<6, 256, 0, stream>>>(Wq, Wk, Wv, vkb, vqb);

  for (int i = 0; i < LNUM; ++i) {
    // z_self = h @ W_self[i,t].T
    gemm(hbfA, WselfB + (size_t)(i * 3 + 0) * D * D, zselfA, NA, D);
    gemm(hbfP, WselfB + (size_t)(i * 3 + 1) * D * D, zselfP, NP, D);
    gemm(hbfC, WselfB + (size_t)(i * 3 + 2) * D * D, zselfC, NC, D);

    // A <- P (ridx 0)
    zero(zrelA, (long)NA * D);
    gemm(hbfP, WsemB + (size_t)(i * 4 + 0) * D * D, tbuf, NP, D);
    scat(ap_row, ap_col, val_ap, tbuf, zrelA, EAP);
    // P <- A (ridx 1)
    zero(zrelP0, (long)NP * D);
    gemm(hbfA, WsemB + (size_t)(i * 4 + 1) * D * D, tbuf, NA, D);
    scat(ap_col, ap_row, val_pa, tbuf, zrelP0, EAP);
    // P <- C (ridx 2)
    zero(zrelP1, (long)NP * D);
    gemm(hbfC, WsemB + (size_t)(i * 4 + 2) * D * D, tbuf, NC, D);
    scat(pc_row, pc_col, val_pc, tbuf, zrelP1, EPC);
    // C <- P (ridx 3)
    zero(zrelC, (long)NC * D);
    gemm(hbfP, WsemB + (size_t)(i * 4 + 3) * D * D, tbuf, NP, D);
    scat(pc_col, pc_row, val_cp, tbuf, zrelC, EPC);

    float* embp = (i == LNUM - 1) ? (float*)d_out : nullptr;
    comb(zselfA, zrelA, nullptr, 1, i * 3 + 0, hbfA, embp, NA);
    comb(zselfP, zrelP0, zrelP1, 2, i * 3 + 1, hbfP, nullptr, NP);
    comb(zselfC, zrelC, nullptr, 1, i * 3 + 2, hbfC, nullptr, NC);
  }

  // predict = softmax(emb @ W_cls.T): GEMM into d_out, softmax in place
  float* predict = (float*)d_out + (size_t)NA * D;
  gemm(hbfA, WclsB, predict, NA, NCLS);
  row_softmax<<<NA, 256, 0, stream>>>(predict, NCLS);
}
